// SparseMoE_25074019074699
// MI455X (gfx1250) — matrix operand order repaired and checked by reading, not compiled, not run on hardware
//
#include <hip/hip_runtime.h>
#include <hip/hip_bf16.h>

typedef __attribute__((ext_vector_type(16))) __bf16 bf16x16;
typedef __attribute__((ext_vector_type(8)))  float  f32x8;

static constexpr int EMBED = 1024;
static constexpr int DFF   = 4096;
static constexpr int NEXP  = 8;
static constexpr int TTOK  = 2048;   // B*S

// ---------------------------------------------------------------------------
// CDNA5 async global->LDS DMA (ASYNCcnt-tracked), per 07_vmem / 08_async_tensor
// ---------------------------------------------------------------------------
__device__ __forceinline__ void async_ld_b128(unsigned lds_off, const void* gaddr) {
    asm volatile("global_load_async_to_lds_b128 %0, %1, off"
                 :: "v"(lds_off), "v"(gaddr) : "memory");
}
__device__ __forceinline__ void wait_async0() {
    asm volatile("s_wait_asynccnt 0x0" ::: "memory");
}

// ---------------------------------------------------------------------------
// fp32 -> bf16 straight convert
// ---------------------------------------------------------------------------
__global__ void cvt_bf16_kernel(const float* __restrict__ in, __bf16* __restrict__ out, int n) {
    int i = blockIdx.x * blockDim.x + threadIdx.x;
    if (i < n) out[i] = (__bf16)in[i];
}

// ---------------------------------------------------------------------------
// fp32 [K][N] -> bf16 [N][K] transpose-convert (32x32 LDS tiles), batched in z
// ---------------------------------------------------------------------------
__global__ __launch_bounds__(256)
void transpose_cvt_kernel(const float* __restrict__ in, __bf16* __restrict__ out, int K, int N) {
    in  += (size_t)blockIdx.z * K * N;
    out += (size_t)blockIdx.z * K * N;
    __shared__ float tile[32][33];
    const int tx = threadIdx.x, ty = threadIdx.y;          // (32, 8)
    const int nb = blockIdx.x * 32, kb = blockIdx.y * 32;
    #pragma unroll
    for (int t = 0; t < 4; ++t)
        tile[ty + t * 8][tx] = in[(size_t)(kb + ty + t * 8) * N + nb + tx];
    __syncthreads();
    #pragma unroll
    for (int t = 0; t < 4; ++t)
        out[(size_t)(nb + ty + t * 8) * K + kb + tx] = (__bf16)tile[tx][ty + t * 8];
}

// ---------------------------------------------------------------------------
// bf16 WMMA GEMM: D[M,N] = op(A[M,K] @ Bt[N,K]^T + bias), optional row
// gather (amap: packed token*2+slot, token = v>>1) and row scatter (dmap).
// Block tile 128x128, 8 waves (4Mx2N), wave tile 32x64 (2x4 of 16x16x32 WMMA).
// Double-buffered LDS filled by GLOBAL_LOAD_ASYNC_TO_LDS_B128; one barrier
// per K-step: wait_asynccnt(0) -> barrier -> issue next tile -> compute.
// ---------------------------------------------------------------------------
template<bool RELU, typename OutT>
__global__ __launch_bounds__(256)
void moe_gemm_bf16(const __bf16* __restrict__ A, const __bf16* __restrict__ Bt,
                   const float* __restrict__ bias, OutT* __restrict__ D,
                   const int* __restrict__ cnt, const int* __restrict__ amap,
                   const int* __restrict__ dmap, int M, int N, int K)
{
    const int nt = cnt ? cnt[0] : M;       // routed-token count (uniform)
    const int m0 = blockIdx.x * 128;
    if (m0 >= nt) return;                  // block-uniform exit only
    const int n0 = blockIdx.y * 128;

    // [buf][ A: 512 chunks (128 rows x 64B) | B: 512 chunks (128 rows x 64B) ]
    __shared__ uint4 smem[2048];           // 32 KB, double buffered

    const int tid  = threadIdx.x;
    const int lane = tid & 31;
    const int wave = tid >> 5;
    const int wm = wave >> 1, wn = wave & 1;   // 4 x 2 wave grid
    const int hf = lane >> 4, lr = lane & 15;

    // global->LDS DMA assignments: 4 chunks of 16B per thread per K-step
    const int rA = tid >> 2;               // A rows rA and rA+64; B rows same
    const int qc = tid & 3;                // 16B chunk within 64B row slice
    const int g0 = m0 + rA, g1 = m0 + rA + 64;
    const bool v0g = g0 < nt, v1g = g1 < nt;
    const int ar0 = v0g ? (amap ? (amap[g0] >> 1) : g0) : 0;
    const int ar1 = v1g ? (amap ? (amap[g1] >> 1) : g1) : 0;
    const uint4* aP0 = (const uint4*)(A + (size_t)ar0 * K);
    const uint4* aP1 = (const uint4*)(A + (size_t)ar1 * K);
    const uint4* bP0 = (const uint4*)(Bt + (size_t)(n0 + rA) * K);
    const uint4* bP1 = (const uint4*)(Bt + (size_t)(n0 + rA + 64) * K);

    f32x8 acc[2][4] = {};

    const int aBase = (wm * 32 + lr) * 4 + hf;            // uint4 index in A half
    const int bBase = 512 + (wn * 64 + lr) * 4 + hf;      // uint4 index in B half

    const int ksteps = K >> 5;

    // issue async DMA for K-step `ks` into buffer `p`
    auto issue = [&](int ks, int p) {
        const int qo = (ks << 2) + qc;     // (k0>>3) + qc with k0 = ks*32
        const unsigned lA = (unsigned)(uintptr_t)&smem[p * 1024 + tid];
        const unsigned lB = (unsigned)(uintptr_t)&smem[p * 1024 + 512 + tid];
        if (v0g) async_ld_b128(lA,              aP0 + qo);
        if (v1g) async_ld_b128(lA + 256 * 16,   aP1 + qo);
        async_ld_b128(lB,            bP0 + qo);
        async_ld_b128(lB + 256 * 16, bP1 + qo);
    };

    issue(0, 0);                           // prologue: tile 0 -> buf 0

    for (int it = 0; it < ksteps; ++it) {
        wait_async0();                     // this wave's DMAs into buf it&1 done
        __syncthreads();                   // all waves' DMAs done; prev buf free
        if (it + 1 < ksteps) issue(it + 1, (it + 1) & 1);

        const int pb = (it & 1) * 1024;
        union { bf16x16 v; uint4 q[2]; } af[2], bfr[4];
        #pragma unroll
        for (int i = 0; i < 2; ++i) {      // A frag: lanes0-15 K[hf*8..], K[16+hf*8..]
            af[i].q[0] = smem[pb + aBase + i * 64];
            af[i].q[1] = smem[pb + aBase + i * 64 + 2];
        }
        #pragma unroll
        for (int j = 0; j < 4; ++j) {      // B frag: lane=col, K[hf*8..], K[16+hf*8..]
            bfr[j].q[0] = smem[pb + bBase + j * 64];
            bfr[j].q[1] = smem[pb + bBase + j * 64 + 2];
        }
        #pragma unroll
        for (int i = 0; i < 2; ++i)
            #pragma unroll
            for (int j = 0; j < 4; ++j)
                acc[i][j] = __builtin_amdgcn_wmma_f32_16x16x32_bf16(
                    false, af[i].v, false, bfr[j].v, (short)0, acc[i][j], false, false);
    }

    #pragma unroll
    for (int i = 0; i < 2; ++i)
        #pragma unroll
        for (int j = 0; j < 4; ++j) {
            const int col = n0 + wn * 64 + j * 16 + lr;
            const float bv = bias[col];
            #pragma unroll
            for (int r = 0; r < 8; ++r) {  // C/D: VGPR r -> M = r + 8*hf
                const int mrow = m0 + wm * 32 + i * 16 + r + hf * 8;
                if (mrow < nt) {
                    float v = acc[i][j][r] + bv;
                    if (RELU) v = fmaxf(v, 0.0f);
                    const int orow = dmap ? dmap[mrow] : mrow;
                    D[(size_t)orow * N + col] = (OutT)v;
                }
            }
        }
}

// ---------------------------------------------------------------------------
// Router second layer: scores[T,8] = H[T,4096]_bf16 @ rw2[4096,8] + rb2.
// One wave per token; cross-lane shuffle reduce. N=8 < 16 -> VALU, not WMMA.
// ---------------------------------------------------------------------------
__global__ __launch_bounds__(256)
void router_scores_kernel(const __bf16* __restrict__ H, const float* __restrict__ rw2,
                          const float* __restrict__ rb2, float* __restrict__ scores)
{
    const int lane = threadIdx.x & 31;
    const int t = blockIdx.x * 8 + (threadIdx.x >> 5);
    float acc[NEXP] = {};
    const __bf16* h = H + (size_t)t * DFF;
    for (int k = lane; k < DFF; k += 32) {
        const float hv = (float)h[k];
        const float4* w = (const float4*)(rw2 + (size_t)k * NEXP);
        const float4 wa = w[0], wb = w[1];
        acc[0] += hv * wa.x; acc[1] += hv * wa.y; acc[2] += hv * wa.z; acc[3] += hv * wa.w;
        acc[4] += hv * wb.x; acc[5] += hv * wb.y; acc[6] += hv * wb.z; acc[7] += hv * wb.w;
    }
    #pragma unroll
    for (int off = 16; off > 0; off >>= 1)
        #pragma unroll
        for (int e = 0; e < NEXP; ++e) acc[e] += __shfl_xor(acc[e], off, 32);
    if (lane == 0) {
        #pragma unroll
        for (int e = 0; e < NEXP; ++e) scores[(size_t)t * NEXP + e] = acc[e] + rb2[e];
    }
}

// ---------------------------------------------------------------------------
// Top-2 + softmax over kept entries; build per-expert compacted token lists.
// ---------------------------------------------------------------------------
__global__ void topk_gate_kernel(const float* __restrict__ scores, float* __restrict__ gatev,
                                 int* __restrict__ counts, int* __restrict__ lists)
{
    const int t = blockIdx.x * blockDim.x + threadIdx.x;
    if (t >= TTOK) return;
    float s[NEXP];
    #pragma unroll
    for (int e = 0; e < NEXP; ++e) s[e] = scores[(size_t)t * NEXP + e];
    int e0 = 0;
    #pragma unroll
    for (int e = 1; e < NEXP; ++e) if (s[e] > s[e0]) e0 = e;
    int e1 = (e0 == 0) ? 1 : 0;
    #pragma unroll
    for (int e = 0; e < NEXP; ++e) if (e != e0 && s[e] > s[e1]) e1 = e;
    const float p1 = __expf(s[e1] - s[e0]);       // s[e0] >= s[e1]
    const float inv = 1.0f / (1.0f + p1);
    gatev[t * 2 + 0] = inv;
    gatev[t * 2 + 1] = p1 * inv;
    int pos = atomicAdd(&counts[e0], 1);
    lists[e0 * TTOK + pos] = t * 2 + 0;
    pos = atomicAdd(&counts[e1], 1);
    lists[e1 * TTOK + pos] = t * 2 + 1;
}

// ---------------------------------------------------------------------------
// out[t,d] = g0 * yslot[t,0,d] + g1 * yslot[t,1,d]
// ---------------------------------------------------------------------------
__global__ void combine_kernel(const float* __restrict__ ys, const float* __restrict__ g,
                               float* __restrict__ out)
{
    const int i = blockIdx.x * blockDim.x + threadIdx.x;
    const int t = i >> 10, d = i & 1023;
    out[i] = g[2 * t] * ys[((size_t)(2 * t) << 10) + d]
           + g[2 * t + 1] * ys[((size_t)(2 * t + 1) << 10) + d];
}

// ---------------------------------------------------------------------------
extern "C" void kernel_launch(void* const* d_in, const int* in_sizes, int n_in,
                              void* d_out, int out_size, void* d_ws, size_t ws_size,
                              hipStream_t stream)
{
    const float* x   = (const float*)d_in[0];
    const float* rw1 = (const float*)d_in[1];
    const float* rb1 = (const float*)d_in[2];
    const float* rw2 = (const float*)d_in[3];
    const float* rb2 = (const float*)d_in[4];
    const float* W1  = (const float*)d_in[5];
    const float* b1  = (const float*)d_in[6];
    const float* W2  = (const float*)d_in[7];
    const float* b2  = (const float*)d_in[8];
    float* out = (float*)d_out;

    char* ws = (char*)d_ws;
    size_t off = 0;
    auto take = [&](size_t bytes) -> char* {
        char* p = ws + off;
        off = (off + bytes + 255) & ~(size_t)255;
        return p;
    };
    __bf16* Xbf    = (__bf16*)take((size_t)TTOK * EMBED * 2);
    __bf16* rw1t   = (__bf16*)take((size_t)EMBED * DFF * 2);
    __bf16* W1t    = (__bf16*)take((size_t)NEXP * EMBED * DFF * 2);
    __bf16* W2t    = (__bf16*)take((size_t)NEXP * EMBED * DFF * 2);
    __bf16* Hbuf   = (__bf16*)take((size_t)TTOK * DFF * 2);
    float*  scores = (float*)take((size_t)TTOK * NEXP * 4);
    float*  gatev  = (float*)take((size_t)TTOK * 2 * 4);
    int*    counts = (int*)take(NEXP * 4);
    int*    lists  = (int*)take((size_t)NEXP * TTOK * 4);
    float*  yslot  = (float*)take((size_t)TTOK * 2 * EMBED * 4);

    hipMemsetAsync(counts, 0, NEXP * sizeof(int), stream);

    // fp32 -> bf16 conversions (weights transposed to [N][K] for B-tiles)
    cvt_bf16_kernel<<<(TTOK * EMBED) / 256, 256, 0, stream>>>(x, Xbf, TTOK * EMBED);
    dim3 tb(32, 8);
    transpose_cvt_kernel<<<dim3(DFF / 32, EMBED / 32, 1),    tb, 0, stream>>>(rw1, rw1t, EMBED, DFF);
    transpose_cvt_kernel<<<dim3(DFF / 32, EMBED / 32, NEXP), tb, 0, stream>>>(W1,  W1t,  EMBED, DFF);
    transpose_cvt_kernel<<<dim3(EMBED / 32, DFF / 32, NEXP), tb, 0, stream>>>(W2,  W2t,  DFF, EMBED);

    // Router hidden: H = relu(X @ rw1 + rb1)
    moe_gemm_bf16<true, __bf16><<<dim3(TTOK / 128, DFF / 128), 256, 0, stream>>>(
        Xbf, rw1t, rb1, Hbuf, nullptr, nullptr, nullptr, TTOK, DFF, EMBED);

    router_scores_kernel<<<TTOK / 8, 256, 0, stream>>>(Hbuf, rw2, rb2, scores);
    topk_gate_kernel<<<TTOK / 256, 256, 0, stream>>>(scores, gatev, counts, lists);

    // Sparse expert MLPs over compacted token lists (Hbuf reused per expert)
    for (int e = 0; e < NEXP; ++e) {
        moe_gemm_bf16<true, __bf16><<<dim3(TTOK / 128, DFF / 128), 256, 0, stream>>>(
            Xbf, W1t + (size_t)e * EMBED * DFF, b1 + (size_t)e * DFF, Hbuf,
            counts + e, lists + (size_t)e * TTOK, nullptr, TTOK, DFF, EMBED);
        moe_gemm_bf16<false, float><<<dim3(TTOK / 128, EMBED / 128), 256, 0, stream>>>(
            Hbuf, W2t + (size_t)e * EMBED * DFF, b2 + (size_t)e * EMBED, yslot,
            counts + e, nullptr, lists + (size_t)e * TTOK, TTOK, EMBED, DFF);
    }

    combine_kernel<<<(TTOK * EMBED) / 256, 256, 0, stream>>>(yslot, gatev, out);
}
